// LinAngularAttention_54211077210524
// MI455X (gfx1250) — matrix operand order repaired and checked by reading, not compiled, not run on hardware
//
#include <hip/hip_runtime.h>

// ---------------------------------------------------------------------------
// LinAngularAttention forward for MI455X (gfx1250, wave32, WMMA bf16).
// ---------------------------------------------------------------------------

#define DEV static __device__ __forceinline__

typedef __bf16 bf16_t;
typedef __attribute__((ext_vector_type(16))) __bf16 bf16x16;
typedef __attribute__((ext_vector_type(8)))  __bf16 bf16x8;
typedef __attribute__((ext_vector_type(8)))  float  f32x8;

// problem constants (from reference setup_inputs)
#define BATCH 8
#define NTOK  4096
#define CH    384
#define NH    8
#define HD    48
#define HDP   64            // head dim padded to 64 so K-loops are 32-aligned
#define C3    1152
#define TOK   (BATCH * NTOK)   // 32768
#define BHP   (BATCH * NH)     // 64 (b,h) pairs
#define KSZ   9
#define INV_PI 0.31830988618379067f

// ---------------------------------------------------------------------------
// small helpers
// ---------------------------------------------------------------------------
DEV bf16_t f2bf(float f) {
    union { float f; unsigned u; } v; v.f = f;
    unsigned r = v.u + 0x7FFFu + ((v.u >> 16) & 1u);   // round-to-nearest-even
    unsigned short h = (unsigned short)(r >> 16);
    bf16_t o; __builtin_memcpy(&o, &h, 2);
    return o;
}
DEV float bf2f(bf16_t b) {
    unsigned short h; __builtin_memcpy(&h, &b, 2);
    union { unsigned u; float f; } v; v.u = ((unsigned)h) << 16;
    return v.f;
}
DEV float wave_sum(float v) {
    #pragma unroll
    for (int off = 16; off; off >>= 1) v += __shfl_xor(v, off, 32);
    return v;
}

// ---------------------------------------------------------------------------
// WMMA fragment loaders (wave32, V_WMMA_F32_16X16X32_BF16)
//
// A (16x32, MxK), row-major [M][K] source, ld = K-stride:
//   lane L (r = L&15, half = L>>4) holds row M=r, K = half*16 + (0..15)
//   -> one contiguous 32-byte load per lane.
// B (32x16, KxN), source is B^T as [N][K] row-major, ld = K-stride:
//   lane L (n = L&15, half = L>>4) holds column N=n, K = half*16 + (0..15)
//   -> one contiguous 32-byte load per lane.
// C/D (16x16 f32): VGPR r holds row (r + half*8), column = lane&15.
// ---------------------------------------------------------------------------
DEV bf16x16 load_fragA(const bf16_t* __restrict__ base, int ld, int lane) {
    const int half = (lane >> 4) & 1;
    const int r    = lane & 15;
    const bf16_t* p = base + (size_t)r * ld + half * 16;
    return *(const bf16x16*)(p);            // K = half*16 + 0..15
}
DEV bf16x16 load_fragB(const bf16_t* __restrict__ baseT, int ld, int lane) {
    const int half = (lane >> 4) & 1;
    const int n    = lane & 15;
    return *(const bf16x16*)(baseT + (size_t)n * ld + half * 16);
}
DEV f32x8 wmma_bf16(bf16x16 a, bf16x16 b, f32x8 c) {
    return __builtin_amdgcn_wmma_f32_16x16x32_bf16(
        /*neg_a=*/false, a, /*neg_b=*/false, b,
        /*c_mod=*/(short)0, c, /*reuse_a=*/false, /*reuse_b=*/false);
}

// ---------------------------------------------------------------------------
// k0: init scalar accumulators
// ---------------------------------------------------------------------------
__global__ void init_scalars(float* s) {
    if (threadIdx.x == 0) { s[0] = 0.f; s[1] = 0.f; s[2] = 0.f; s[3] = 0.f; }
}

// ---------------------------------------------------------------------------
// k1: f32 -> bf16 conversion
// ---------------------------------------------------------------------------
__global__ void cvt_f32_bf16(const float* __restrict__ src, bf16_t* __restrict__ dst, int n) {
    int i = blockIdx.x * blockDim.x + threadIdx.x;
    if (i < n) dst[i] = f2bf(src[i]);
}

// ---------------------------------------------------------------------------
// k2: top-1-of-2 gate (one wave per token); argmax ties -> expert 0 (jnp.argmax)
// ---------------------------------------------------------------------------
__global__ void gate_argmax(const bf16_t* __restrict__ X, const float* __restrict__ gw,
                            const float* __restrict__ gb, int* __restrict__ sel, int K) {
    const int t = blockIdx.x, lane = threadIdx.x;
    float l0 = 0.f, l1 = 0.f;
    for (int k = lane; k < K; k += 32) {
        float xv = bf2f(X[(size_t)t * K + k]);
        l0 += xv * gw[k * 2 + 0];
        l1 += xv * gw[k * 2 + 1];
    }
    l0 = wave_sum(l0); l1 = wave_sum(l1);
    if (lane == 0) sel[t] = ((l1 + gb[1]) > (l0 + gb[0])) ? 1 : 0;
}

// ---------------------------------------------------------------------------
// k3/k9: MoE GEMM  Y[t,n] = X[t,:] . W_sel[t][n,:] + b_sel[t][n]
// Both experts computed with WMMA, per-row select at store (bandwidth-bound
// workload: the extra FLOPs are free on MI455X).
// One wave per 16x16 output tile. grid = (T/16, Nout/16).
// ---------------------------------------------------------------------------
__global__ void moe_gemm_wmma(const bf16_t* __restrict__ X,
                              const bf16_t* __restrict__ W0, const bf16_t* __restrict__ W1,
                              const float* __restrict__ bias0, const float* __restrict__ bias1,
                              const int* __restrict__ sel, float* __restrict__ Y,
                              int K, int Nout) {
    const int lane = threadIdx.x & 31;
    const int m0 = blockIdx.x * 16;
    const int n0 = blockIdx.y * 16;
    const bf16_t* Arow  = X  + (size_t)m0 * K;
    const bf16_t* B0row = W0 + (size_t)n0 * K;   // W is [Nout][K] row-major == B^T
    const bf16_t* B1row = W1 + (size_t)n0 * K;
    f32x8 acc0 = {}, acc1 = {};
    for (int k0 = 0; k0 < K; k0 += 32) {
        __builtin_prefetch(Arow + k0 + 128, 0, 1);   // global_prefetch_b8
        bf16x16 a  = load_fragA(Arow  + k0, K, lane);
        bf16x16 b0 = load_fragB(B0row + k0, K, lane);
        bf16x16 b1 = load_fragB(B1row + k0, K, lane);
        acc0 = wmma_bf16(a, b0, acc0);
        acc1 = wmma_bf16(a, b1, acc1);
    }
    const int half = lane >> 4, n = lane & 15;
    const float bb0 = bias0 ? bias0[n0 + n] : 0.f;
    const float bb1 = bias1 ? bias1[n0 + n] : 0.f;
    #pragma unroll
    for (int r = 0; r < 8; ++r) {
        int row = m0 + r + half * 8;
        int s = sel[row];
        Y[(size_t)row * Nout + n0 + n] = s ? (acc1[r] + bb1) : (acc0[r] + bb0);
    }
}

// ---------------------------------------------------------------------------
// k4: per-row reduction of |q|/||q|| and |k|/||k|| -> global scale sums.
// Note (q/||q|| > 0) <=> (q > 0), so normalized q/k never materialize.
// grid = (NTOK, BH), one wave per (b,h,n) row.
// ---------------------------------------------------------------------------
__global__ void qk_scale_reduce(const float* __restrict__ qkv, float* __restrict__ sums) {
    const int n = blockIdx.x, bh = blockIdx.y, lane = threadIdx.x;
    const int b = bh >> 3, h = bh & 7;
    const float* row = qkv + ((size_t)(b * NTOK + n)) * C3 + h * HD;
    float qs = 0.f, qa = 0.f, ks = 0.f, ka = 0.f;
    for (int e = lane; e < HD; e += 32) {
        float qv = row[e], kv = row[CH + e];
        qs += qv * qv; qa += fabsf(qv);
        ks += kv * kv; ka += fabsf(kv);
    }
    qs = wave_sum(qs); qa = wave_sum(qa);
    ks = wave_sum(ks); ka = wave_sum(ka);
    if (lane == 0) {
        atomicAdd(&sums[0], qa * rsqrtf(qs));
        atomicAdd(&sums[1], ka * rsqrtf(ks));
    }
}
__global__ void finalize_scales(float* s) {
    const float inv = 1.0f / ((float)BHP * (float)NTOK * (float)HD);
    s[2] = s[0] * inv;   // scale1 = mean(|q_norm|)
    s[3] = s[1] * inv;   // scale2 = mean(|k_norm|)
}

// ---------------------------------------------------------------------------
// k5: binarize + pack (d padded to 64 with zeros):
//   qb [bh][n][64]   = (q>0)?scale1:0       (A operand of Q.attn GEMM)
//   kT [bh][64][n]   = (k>0)?scale2:0       (A operand of K^T.V GEMM, row=d)
//   vT [bh][64][n]   = v                    (B^T operand of K^T.V GEMM)
// flat grid over BH*NTOK*64 elements.
// ---------------------------------------------------------------------------
__global__ void binarize_pack(const float* __restrict__ qkv, const float* __restrict__ scales,
                              bf16_t* __restrict__ qb, bf16_t* __restrict__ kT,
                              bf16_t* __restrict__ vT) {
    const long i = (long)blockIdx.x * blockDim.x + threadIdx.x;
    const int e  = (int)(i & 63);
    const int n  = (int)((i >> 6) & (NTOK - 1));
    const int bh = (int)(i >> 18);
    const int b = bh >> 3, h = bh & 7;
    const float* row = qkv + ((size_t)(b * NTOK + n)) * C3 + h * HD;
    float qv = 0.f, kv = 0.f, vv = 0.f;
    if (e < HD) {
        qv = (row[e]          > 0.f) ? scales[2] : 0.f;
        kv = (row[CH + e]     > 0.f) ? scales[3] : 0.f;
        vv =  row[2 * CH + e];
    }
    const size_t base = (size_t)bh * NTOK * HDP;
    qb[base + (size_t)n * HDP + e]  = f2bf(qv);
    kT[base + (size_t)e * NTOK + n] = f2bf(kv);
    vT[base + (size_t)e * NTOK + n] = f2bf(vv);
}

// ---------------------------------------------------------------------------
// k6: attn[d,e] = sum_n k[n,d] v[n,e]  (per (b,h); M=64, N=64, K=4096)
// A = kT rows (contiguous over n), B^T = vT rows (contiguous over n).
// Result stored *transposed* as bf16 attnT[bh][e][d] so it is directly the
// B^T operand of the next GEMM. grid = (4, 4, 64), one wave per tile.
// ---------------------------------------------------------------------------
__global__ void attn_kv_wmma(const bf16_t* __restrict__ kT, const bf16_t* __restrict__ vT,
                             bf16_t* __restrict__ attnT) {
    const int lane = threadIdx.x & 31;
    const int dt = blockIdx.x, et = blockIdx.y, bh = blockIdx.z;
    const size_t base = (size_t)bh * NTOK * HDP;
    const bf16_t* A  = kT + base + (size_t)dt * 16 * NTOK;
    const bf16_t* Bt = vT + base + (size_t)et * 16 * NTOK;
    f32x8 acc = {};
    for (int k0 = 0; k0 < NTOK; k0 += 32) {
        __builtin_prefetch(A + k0 + 256, 0, 1);
        __builtin_prefetch(Bt + k0 + 256, 0, 1);
        acc = wmma_bf16(load_fragA(A + k0, NTOK, lane),
                        load_fragB(Bt + k0, NTOK, lane), acc);
    }
    const int half = lane >> 4, n = lane & 15;   // n = e within tile
    bf16_t* out = attnT + (size_t)bh * HDP * HDP + (size_t)(et * 16 + n) * HDP + dt * 16;
    #pragma unroll
    for (int r = 0; r < 8; ++r) out[r + half * 8] = f2bf(acc[r]);   // [e][d]
}

// ---------------------------------------------------------------------------
// k7: out_lin[n,e] = sum_d qb[n,d] * attn[d,e]   (M=4096, N=48, K=64)
// A = qb rows, B^T = attnT rows. grid = (256, 3, 64).
// ---------------------------------------------------------------------------
__global__ void q_attn_wmma(const bf16_t* __restrict__ qb, const bf16_t* __restrict__ attnT,
                            float* __restrict__ out_lin) {
    const int lane = threadIdx.x & 31;
    const int mt = blockIdx.x, et = blockIdx.y, bh = blockIdx.z;
    const bf16_t* A  = qb    + (size_t)bh * NTOK * HDP + (size_t)mt * 16 * HDP;
    const bf16_t* Bt = attnT + (size_t)bh * HDP * HDP  + (size_t)et * 16 * HDP;
    f32x8 acc = {};
    #pragma unroll
    for (int k0 = 0; k0 < HDP; k0 += 32)
        acc = wmma_bf16(load_fragA(A + k0, HDP, lane),
                        load_fragB(Bt + k0, HDP, lane), acc);
    const int half = lane >> 4, n = lane & 15;
    float* out = out_lin + ((size_t)bh * NTOK + (size_t)mt * 16) * HD + et * 16 + n;
    #pragma unroll
    for (int r = 0; r < 8; ++r) out[(size_t)(r + half * 8) * HD] = acc[r];
}

// ---------------------------------------------------------------------------
// k8: combine: t = 0.5*v - (1/pi)*out_lin ; t /= ||t|| ; t += depthwise conv(v)
// then transpose-write to out_mid[b][n][h*48+e] as bf16 for the proj GEMM.
// One wave per (b,h,n); lane handles e=lane and e=lane+32.
// ---------------------------------------------------------------------------
__global__ void combine_norm_dconv(const float* __restrict__ qkv, const float* __restrict__ out_lin,
                                   const float* __restrict__ dw, bf16_t* __restrict__ out_mid) {
    const int n = blockIdx.x, bh = blockIdx.y, lane = threadIdx.x;
    const int b = bh >> 3, h = bh & 7;
    const float* lin = out_lin + ((size_t)bh * NTOK + n) * HD;
    const float* vbase = qkv + (size_t)(b * NTOK) * C3 + 2 * CH + h * HD;  // + n*C3 + e
    float val[2]; float ss = 0.f;
    #pragma unroll
    for (int j = 0; j < 2; ++j) {
        const int e = lane + 32 * j;
        float t = 0.f;
        if (e < HD) {
            float vv = vbase[(size_t)n * C3 + e];
            t = 0.5f * vv - INV_PI * lin[e];
            ss += t * t;
        }
        val[j] = t;
    }
    const float rn = rsqrtf(wave_sum(ss));
    bf16_t* orow = out_mid + ((size_t)(b * NTOK + n)) * CH + h * HD;
    #pragma unroll
    for (int j = 0; j < 2; ++j) {
        const int e = lane + 32 * j;
        if (e < HD) {
            float dc = 0.f;
            #pragma unroll
            for (int t9 = 0; t9 < KSZ; ++t9) {
                int nn = n + t9 - KSZ / 2;
                if (nn >= 0 && nn < NTOK)
                    dc += dw[h * KSZ + t9] * vbase[(size_t)nn * C3 + e];
            }
            orow[e] = f2bf(val[j] * rn + dc);
        }
    }
}

// ---------------------------------------------------------------------------
// host: workspace carving + launch sequence (graph-capture safe: no mallocs)
// ---------------------------------------------------------------------------
extern "C" void kernel_launch(void* const* d_in, const int* in_sizes, int n_in,
                              void* d_out, int out_size, void* d_ws, size_t ws_size,
                              hipStream_t stream) {
    (void)in_sizes; (void)n_in; (void)out_size; (void)ws_size;
    const float* x          = (const float*)d_in[0];
    // d_in[1], d_in[2] = H, W (unused by the math)
    const float* qkv_gate_w = (const float*)d_in[3];
    const float* qkv_gate_b = (const float*)d_in[4];
    const float* qkv_w0     = (const float*)d_in[5];
    const float* qkv_w1     = (const float*)d_in[6];
    const float* proj_gate_w= (const float*)d_in[7];
    const float* proj_gate_b= (const float*)d_in[8];
    const float* proj_w0    = (const float*)d_in[9];
    const float* proj_b0    = (const float*)d_in[10];
    const float* proj_w1    = (const float*)d_in[11];
    const float* proj_b1    = (const float*)d_in[12];
    const float* dconv_w    = (const float*)d_in[13];
    float* out = (float*)d_out;

    // -------- workspace layout (256B aligned chunks) --------
    char* ws = (char*)d_ws;
    size_t off = 0;
    auto carve = [&](size_t bytes) -> void* {
        void* p = ws + off;
        off += (bytes + 255) & ~(size_t)255;
        return p;
    };
    float*  scal     = (float*)carve(4 * sizeof(float));
    bf16_t* x_bf     = (bf16_t*)carve((size_t)TOK * CH * 2);       // reused as out_mid
    bf16_t* w0q_bf   = (bf16_t*)carve((size_t)C3 * CH * 2);
    bf16_t* w1q_bf   = (bf16_t*)carve((size_t)C3 * CH * 2);
    bf16_t* w0p_bf   = (bf16_t*)carve((size_t)CH * CH * 2);
    bf16_t* w1p_bf   = (bf16_t*)carve((size_t)CH * CH * 2);
    int*    sel1     = (int*)carve((size_t)TOK * 4);
    int*    sel2     = (int*)carve((size_t)TOK * 4);
    float*  qkv_f32  = (float*)carve((size_t)TOK * C3 * 4);
    bf16_t* qb       = (bf16_t*)carve((size_t)BHP * NTOK * HDP * 2);
    bf16_t* kT       = (bf16_t*)carve((size_t)BHP * NTOK * HDP * 2);
    bf16_t* vT       = (bf16_t*)carve((size_t)BHP * NTOK * HDP * 2);
    bf16_t* attnT    = (bf16_t*)carve((size_t)BHP * HDP * HDP * 2);
    float*  out_lin  = (float*)carve((size_t)BHP * NTOK * HD * 4);
    bf16_t* out_mid  = x_bf;   // x_bf dead after the qkv GEMM

    // 0) zero global scale accumulators
    init_scalars<<<1, 32, 0, stream>>>(scal);

    // 1) f32 -> bf16 packs
    {
        int n = TOK * CH;
        cvt_f32_bf16<<<(n + 255) / 256, 256, 0, stream>>>(x, x_bf, n);
        n = C3 * CH;
        cvt_f32_bf16<<<(n + 255) / 256, 256, 0, stream>>>(qkv_w0, w0q_bf, n);
        cvt_f32_bf16<<<(n + 255) / 256, 256, 0, stream>>>(qkv_w1, w1q_bf, n);
        n = CH * CH;
        cvt_f32_bf16<<<(n + 255) / 256, 256, 0, stream>>>(proj_w0, w0p_bf, n);
        cvt_f32_bf16<<<(n + 255) / 256, 256, 0, stream>>>(proj_w1, w1p_bf, n);
    }

    // 2) qkv gate + MoE GEMM: [32768,384] x [384,1152] (both experts, WMMA)
    gate_argmax<<<TOK, 32, 0, stream>>>(x_bf, qkv_gate_w, qkv_gate_b, sel1, CH);
    moe_gemm_wmma<<<dim3(TOK / 16, C3 / 16), 32, 0, stream>>>(
        x_bf, w0q_bf, w1q_bf, nullptr, nullptr, sel1, qkv_f32, CH, C3);

    // 3) global binarization scales, then pack q/k/v (d padded to 64)
    qk_scale_reduce<<<dim3(NTOK, BHP), 32, 0, stream>>>(qkv_f32, scal);
    finalize_scales<<<1, 1, 0, stream>>>(scal);
    {
        long total = (long)BHP * NTOK * HDP;
        binarize_pack<<<(unsigned)(total / 256), 256, 0, stream>>>(qkv_f32, scal, qb, kT, vT);
    }

    // 4) linear-attention reorder: attn = K^T V (K=4096), then Q . attn (K=64)
    attn_kv_wmma<<<dim3(HDP / 16, HDP / 16, BHP), 32, 0, stream>>>(kT, vT, attnT);
    q_attn_wmma<<<dim3(NTOK / 16, HD / 16, BHP), 32, 0, stream>>>(qb, attnT, out_lin);

    // 5) 0.5*v - (1/pi)*QA, row-normalize, + depthwise conv, transpose to [B,N,C]
    combine_norm_dconv<<<dim3(NTOK, BHP), 32, 0, stream>>>(qkv_f32, out_lin, dconv_w, out_mid);

    // 6) proj gate + MoE GEMM: [32768,384] x [384,384] -> d_out (f32)
    gate_argmax<<<TOK, 32, 0, stream>>>(out_mid, proj_gate_w, proj_gate_b, sel2, CH);
    moe_gemm_wmma<<<dim3(TOK / 16, CH / 16), 32, 0, stream>>>(
        out_mid, w0p_bf, w1p_bf, proj_b0, proj_b1, sel2, out, CH, CH);
}
